// AttentionGatedCache_46162308497809
// MI455X (gfx1250) — matrix operand order repaired and checked by reading, not compiled, not run on hardware
//
#include <hip/hip_runtime.h>

// PolarQuant-style quantize/dequantize round trip for MI455X (gfx1250, wave32).
//
// Roofline: 268 MB HBM traffic -> ~11.5 us floor @ 23.3 TB/s; 17.2 GFLOP of
// GEMM work. We use V_WMMA_F32_16X16X32_F16 with a hi/lo f16 error-split
// (3 WMMAs per product: hi*Rhi + hi*Rlo + lo*Rhi) to get ~fp32 accuracy for
// the rotation while staying near the memory roof (fp32 16x16x4 WMMA would be
// ~8x more instructions per FLOP).

typedef __attribute__((ext_vector_type(16))) _Float16 v16h;
typedef __attribute__((ext_vector_type(2)))  _Float16 h2v;
typedef __attribute__((ext_vector_type(8)))  float    v8f;

#define DDIM        128
#define ROWS_PER_WV 16
#define WAVES_PER_B 8
#define AROW        132   // f16 stride of A-staging rows (pad vs 64-bank LDS)
#define XROW        132   // f32 stride of xr-staging rows
#define CROW        132   // f32 stride of centroid rows

// Dynamic LDS layout (bytes):
//  [0)        B fragments: 4 sets (B1hi,B1lo,B2hi,B2lo) x 32 frags x 512 f16
#define OFF_BF    0
#define NFRAGH    16384            // f16 per set
//  [131072)   per-wave A staging: hi[16*132] + lo[16*132] f16
#define OFF_AST   131072
#define AST_WV    (2 * ROWS_PER_WV * AROW)   // 4224 f16 per wave
//  [198656)   per-wave xr staging: 16*132 f32
#define OFF_XR    198656
#define XR_WV     (ROWS_PER_WV * XROW)       // 2112 f32 per wave
//  [266240)   centroids 10*132 f32 + 0.5*||c||^2 [10]
#define OFF_CENT  266240
#define SMEM_SZ   271616

// ---------------------------------------------------------------------------
// Prep: swizzle R (and R^T) into WMMA B-matrix fragments, split hi/lo f16.
// B k order used here (contiguous, same as A): lane = {half: K+=16, low4: N}; element e -> K = e.
// Fragment id f = nt*4 + kt; per-lane 16 f16 stored contiguously (32B loads).
// ---------------------------------------------------------------------------
__global__ void pq_prep_frags(const float* __restrict__ R,
                              _Float16* __restrict__ ws) {
  int idx = blockIdx.x * 256 + threadIdx.x;   // 0 .. 16383
  if (idx >= 32 * 512) return;
  int frag = idx >> 9;
  int lane = (idx >> 4) & 31;
  int e    = idx & 15;
  int nt   = frag >> 2;
  int kt   = frag & 3;
  int k = kt * 32 + e + ((lane >> 4) << 4);
  int n = nt * 16 + (lane & 15);
  float v1 = R[k * DDIM + n];      // B1 = R      (for  xn @ R)
  float v2 = R[n * DDIM + k];      // B2 = R^T    (for  combined @ R^T)
  _Float16 h1 = (_Float16)v1;
  _Float16 l1 = (_Float16)(v1 - (float)h1);
  _Float16 h2 = (_Float16)v2;
  _Float16 l2 = (_Float16)(v2 - (float)h2);
  ws[idx]             = h1;
  ws[NFRAGH + idx]    = l1;
  ws[2*NFRAGH + idx]  = h2;
  ws[3*NFRAGH + idx]  = l2;
}

// ---------------------------------------------------------------------------
// One 16x128 @ 128x128 GEMM tile with error-split f16 WMMA.
// A k order used here (contiguous, same as B): lane%16 = M; element e -> K = e +
// (lane>>4)*16.
// ---------------------------------------------------------------------------
__device__ __forceinline__ void pq_gemm_tile(const _Float16* __restrict__ Ah,
                                             const _Float16* __restrict__ Al,
                                             const _Float16* __restrict__ Bh,
                                             const _Float16* __restrict__ Bl,
                                             int lane, v8f* acc) {
  const int m     = lane & 15;
  const int half8 = (lane >> 4) << 3;
#pragma unroll
  for (int kt = 0; kt < 4; ++kt) {
    v16h ahi, alo;
#pragma unroll
    for (int j = 0; j < 8; ++j) {
      const int kk = kt * 32 + (j << 1) + (half8 << 1);
      h2v ph = *(const h2v*)&Ah[m * AROW + kk];
      h2v pl = *(const h2v*)&Al[m * AROW + kk];
      ahi[2 * j]     = ph[0];
      ahi[2 * j + 1] = ph[1];
      alo[2 * j]     = pl[0];
      alo[2 * j + 1] = pl[1];
    }
#pragma unroll
    for (int nt = 0; nt < 8; ++nt) {
      const int f = (nt * 4 + kt) * 512 + lane * 16;
      v16h bhi = *(const v16h*)&Bh[f];
      v16h blo = *(const v16h*)&Bl[f];
      acc[nt] = __builtin_amdgcn_wmma_f32_16x16x32_f16(
          false, ahi, false, bhi, (short)0, acc[nt], false, false);
      acc[nt] = __builtin_amdgcn_wmma_f32_16x16x32_f16(
          false, ahi, false, blo, (short)0, acc[nt], false, false);
      acc[nt] = __builtin_amdgcn_wmma_f32_16x16x32_f16(
          false, alo, false, bhi, (short)0, acc[nt], false, false);
    }
  }
}

// ---------------------------------------------------------------------------
// Main kernel: 256 threads = 8 wave32, 16 rows per wave, 128 rows per block.
// ---------------------------------------------------------------------------
__global__ __launch_bounds__(256) void pq_main(
    const float* __restrict__ x, const unsigned char* __restrict__ rmask,
    const float* __restrict__ bc, const float* __restrict__ rc,
    const _Float16* __restrict__ frags, float* __restrict__ out) {
  extern __shared__ char smem[];
  _Float16* Bf    = (_Float16*)(smem + OFF_BF);
  _Float16* Asta  = (_Float16*)(smem + OFF_AST);
  float*    XRs   = (float*)(smem + OFF_XR);
  float*    cent  = (float*)(smem + OFF_CENT);
  float*    centh = cent + 10 * CROW;

  const int tid = threadIdx.x;

  // ---- phase 0: stage B fragments + centroids into LDS --------------------
  {
    const uint4* src = (const uint4*)frags;
    uint4*       dst = (uint4*)Bf;
    for (int i = tid; i < (4 * NFRAGH) / 8; i += 256) dst[i] = src[i];
    for (int i = tid; i < 10 * DDIM; i += 256) {
      int c = i >> 7, j = i & 127;
      float v = (c < 2) ? bc[c * DDIM + j] : rc[(c - 2) * DDIM + j];
      cent[c * CROW + j] = v;
    }
    if (tid < 10) {
      const float* s = (tid < 2) ? (bc + tid * DDIM) : (rc + (tid - 2) * DDIM);
      float a = 0.f;
      for (int j = 0; j < DDIM; ++j) a += s[j] * s[j];
      centh[tid] = 0.5f * a;
    }
  }
  __syncthreads();

  const int w        = tid >> 5;
  const int lane     = tid & 31;
  const int m        = lane & 15;
  const int hsel     = lane >> 4;
  const int tileBase = blockIdx.x * (WAVES_PER_B * ROWS_PER_WV) + w * ROWS_PER_WV;

  _Float16* Ah = Asta + w * AST_WV;
  _Float16* Al = Ah + ROWS_PER_WV * AROW;
  float*    xr = XRs + w * XR_WV;

  // ---- phase 1: load 16 rows coalesced, norms via shfl, hi/lo f16 to LDS --
  float norms[16];
  {
    const float4* xg = (const float4*)(x + (size_t)tileBase * DDIM);
#pragma unroll
    for (int i = 0; i < 16; ++i) {
      float4 v = xg[i * 32 + lane];
      float ss = v.x * v.x + v.y * v.y + v.z * v.z + v.w * v.w;
      ss += __shfl_xor(ss, 1);
      ss += __shfl_xor(ss, 2);
      ss += __shfl_xor(ss, 4);
      ss += __shfl_xor(ss, 8);
      ss += __shfl_xor(ss, 16);
      norms[i] = sqrtf(ss);
      float inv = 1.0f / (norms[i] + 1e-8f);
      float a0 = v.x * inv, a1 = v.y * inv, a2 = v.z * inv, a3 = v.w * inv;
      _Float16 f0 = (_Float16)a0, f1 = (_Float16)a1;
      _Float16 f2 = (_Float16)a2, f3 = (_Float16)a3;
      int off = i * AROW + 4 * lane;
      h2v hp0 = {f0, f1}, hp1 = {f2, f3};
      *(h2v*)&Ah[off]     = hp0;
      *(h2v*)&Ah[off + 2] = hp1;
      h2v lp0 = {(_Float16)(a0 - (float)f0), (_Float16)(a1 - (float)f1)};
      h2v lp1 = {(_Float16)(a2 - (float)f2), (_Float16)(a3 - (float)f3)};
      *(h2v*)&Al[off]     = lp0;
      *(h2v*)&Al[off + 2] = lp1;
    }
  }
  __syncthreads();

  // ---- phase 2: xr = xn @ R ----------------------------------------------
  v8f acc[8];
  const v8f vzero = {0.f, 0.f, 0.f, 0.f, 0.f, 0.f, 0.f, 0.f};
#pragma unroll
  for (int nt = 0; nt < 8; ++nt) acc[nt] = vzero;
  pq_gemm_tile(Ah, Al, Bf, Bf + NFRAGH, lane, acc);

  // ---- phase 3: spill xr (D-layout: m = r + 8*hsel, n = nt*16 + m) --------
#pragma unroll
  for (int nt = 0; nt < 8; ++nt)
#pragma unroll
    for (int r = 0; r < 8; ++r)
      xr[(r + hsel * 8) * XROW + nt * 16 + m] = acc[nt][r];
  __syncthreads();

  // ---- phase 4: per-row quantize/encode/select (row m, half hsel) ---------
  {
    float v[64];
    const float4* row4 = (const float4*)(xr + m * XROW + hsel * 64);
#pragma unroll
    for (int t = 0; t < 16; ++t) {
      float4 q = row4[t];
      v[4 * t] = q.x; v[4 * t + 1] = q.y; v[4 * t + 2] = q.z; v[4 * t + 3] = q.w;
    }
    float bestb = -3.4e38f, bestr = -3.4e38f;
    int bi = 0, ri = 2;
#pragma unroll
    for (int c = 0; c < 10; ++c) {
      const float4* cc = (const float4*)(cent + c * CROW + hsel * 64);
      float p = 0.f;
#pragma unroll
      for (int t = 0; t < 16; ++t) {
        float4 q = cc[t];
        p += v[4*t] * q.x + v[4*t+1] * q.y + v[4*t+2] * q.z + v[4*t+3] * q.w;
      }
      p += __shfl_xor(p, 16);
      float sc = p - centh[c];
      if (c < 2) { if (sc > bestb) { bestb = sc; bi = c; } }
      else       { if (sc > bestr) { bestr = sc; ri = c; } }
    }
    const int sel = (rmask[tileBase + m] != 0) ? ri : bi;
    const float* cs = cent + sel * CROW + hsel * 64;
    float s = 0.f;
#pragma unroll
    for (int t = 0; t < 64; ++t) s += fabsf(v[t] - cs[t]);
    s += __shfl_xor(s, 16);
    const float scale = s * (1.0f / 128.0f);
#pragma unroll
    for (int t = 0; t < 64; t += 2) {
      float c0 = cs[t]     + ((v[t]     - cs[t])     >= 0.f ? scale : -scale);
      float c1 = cs[t + 1] + ((v[t + 1] - cs[t + 1]) >= 0.f ? scale : -scale);
      _Float16 f0 = (_Float16)c0, f1 = (_Float16)c1;
      int off = m * AROW + hsel * 64 + t;
      h2v hp = {f0, f1};
      *(h2v*)&Ah[off] = hp;
      h2v lp = {(_Float16)(c0 - (float)f0), (_Float16)(c1 - (float)f1)};
      *(h2v*)&Al[off] = lp;
    }
  }
  __syncthreads();

  // ---- phase 5: x_recon = combined @ R^T, rescale by norm, store ----------
#pragma unroll
  for (int nt = 0; nt < 8; ++nt) acc[nt] = vzero;
  pq_gemm_tile(Ah, Al, Bf + 2 * NFRAGH, Bf + 3 * NFRAGH, lane, acc);

#pragma unroll
  for (int nt = 0; nt < 8; ++nt)
#pragma unroll
    for (int r = 0; r < 8; ++r) {
      float nm = hsel ? norms[r + 8] : norms[r];
      int mrow = r + hsel * 8;
      out[(size_t)(tileBase + mrow) * DDIM + nt * 16 + m] = acc[nt][r] * nm;
    }
}

// ---------------------------------------------------------------------------
// Host entry.
// Inputs: [0] x (f32, N*128)  [1] refine_mask (bool, N)  [2] R (f32, 128*128)
//         [3] base_centroids (f32, 2*128)  [4] refine_centroids (f32, 8*128)
// Output: f32, N*128. d_ws holds the pre-swizzled B fragments (128 KB).
// ---------------------------------------------------------------------------
extern "C" void kernel_launch(void* const* d_in, const int* in_sizes, int n_in,
                              void* d_out, int out_size, void* d_ws,
                              size_t ws_size, hipStream_t stream) {
  const float*         xp    = (const float*)d_in[0];
  const unsigned char* maskp = (const unsigned char*)d_in[1];  // jax bool = 1B
  const float*         Rp    = (const float*)d_in[2];
  const float*         bcp   = (const float*)d_in[3];
  const float*         rcp   = (const float*)d_in[4];
  float*               outp  = (float*)d_out;
  _Float16*            frags = (_Float16*)d_ws;

  const int N       = in_sizes[0] / DDIM;                       // 262144
  const int nblocks = N / (WAVES_PER_B * ROWS_PER_WV);          // 2048

  // Opt in to >64KB dynamic LDS (320 KB/WGP on gfx1250).
  (void)hipFuncSetAttribute((const void*)pq_main,
                            hipFuncAttributeMaxDynamicSharedMemorySize,
                            SMEM_SZ);

  pq_prep_frags<<<64, 256, 0, stream>>>(Rp, frags);
  pq_main<<<nblocks, 256, SMEM_SZ, stream>>>(xp, maskp, bcp, rcp, frags, outp);
}
